// WarpBlock_36412732735527
// MI455X (gfx1250) — compile-verified, matrix operand order checked by reading, not run on hardware
//
#include <hip/hip_runtime.h>
#include <hip/hip_bf16.h>
#include <cmath>

// CDNA5 / gfx1250 WMMA implementation of the DCNv2 "WarpBlock".
//
// Layout-transformed pipeline:
//   pack_cat : concat(x_vq,x_res) fp32 NCHW -> bf16 NHWC [B,H,W,128]
//   pack_w   : weights -> bf16 B-fragment order [tap][g][nt][lane][8dw]
//   conv1    : NHWC-128 x packed-W1 -> feat bf16 NHWC [B,H,W,64]
//   conv2/3  : NHWC-64  x packed-W  -> offset / mod, fp32 NCHW
//   dcn      : bilinear-sampled feat (mod folded into corner weights)
//              x packed-Wdcn -> out fp32 NCHW
//
// Every A/B fragment is fetched with 2x global_load_b128 (no per-element
// cvts in the hot loops); zero-padding & partial N-tiles are branchless
// bitwise masks.  Epilogues transpose through LDS for coalesced stores.

typedef __attribute__((ext_vector_type(16))) __bf16   v16bf;
typedef __attribute__((ext_vector_type(8)))  float    v8f;
typedef __attribute__((ext_vector_type(8)))  unsigned v8u;

namespace {
constexpr int Bn = 8, Cc = 64, Hh = 128, Ww = 128;
constexpr int PLANE      = Hh * Ww;
constexpr int MT_PER_IMG = Hh * (Ww / 16);   // 1024 16-pixel tiles per image
constexpr int MTILES     = Bn * MT_PER_IMG;  // 8192 M-tiles total
}

// K-slot -> K index inside a 16x32 bf16 A/B fragment (ISA 7.12.2 layout).
__device__ __forceinline__ constexpr int kslot(int s, int half) {
  int v = s >> 1;
  return ((v >> 2) << 4) + (half << 3) + ((v & 3) << 1) + (s & 1);
}

__device__ __forceinline__ unsigned pk_bf16(float a, float b) {
  return (unsigned)__builtin_bit_cast(unsigned short, (__bf16)a) |
         ((unsigned)__builtin_bit_cast(unsigned short, (__bf16)b) << 16);
}

// Branchless lane-uniform zeroing of a bf16 fragment (8x v_and_b32).
__device__ __forceinline__ v8u mask8(v8u u, unsigned m) {
#pragma unroll
  for (int i = 0; i < 8; ++i) u[i] &= m;
  return u;
}

// Two contiguous 16B runs -> one 16x32 A/B fragment (per-lane 8 dwords).
__device__ __forceinline__ v8u load_frag_runs(const __bf16* p) {
  uint4 lo = *reinterpret_cast<const uint4*>(p);        // ch +0..7
  uint4 hi = *reinterpret_cast<const uint4*>(p + 16);   // ch +16..23
  return (v8u){lo.x, lo.y, lo.z, lo.w, hi.x, hi.y, hi.z, hi.w};
}

// ---------------------------------------------------------------------------
// One-shot layout packers.
// ---------------------------------------------------------------------------
__global__ __launch_bounds__(256)
void pack_cat(const float* __restrict__ x0, const float* __restrict__ x1,
              unsigned* __restrict__ cat /* [B*H*W][64 dwords] */) {
  const int p  = blockIdx.x * 256 + threadIdx.x;        // pixel id
  const int b  = p / PLANE, pp = p % PLANE;
  const float* s0 = x0 + (size_t)b * 64 * PLANE + pp;
  const float* s1 = x1 + (size_t)b * 64 * PLANE + pp;
  unsigned* dst = cat + (size_t)p * 64;
#pragma unroll
  for (int c2 = 0; c2 < 32; ++c2)
    dst[c2] = pk_bf16(s0[(2 * c2) * PLANE], s0[(2 * c2 + 1) * PLANE]);
#pragma unroll
  for (int c2 = 0; c2 < 32; ++c2)
    dst[32 + c2] = pk_bf16(s1[(2 * c2) * PLANE], s1[(2 * c2 + 1) * PLANE]);
}

// Weights OIHW fp32 -> packed B fragments: dword index
//   (((tap*G + g)*NT + nt)*32 + lane)*8 + v
template <int CIN, int COUT, int NT>
__global__ __launch_bounds__(256)
void pack_w(const float* __restrict__ w, unsigned* __restrict__ wp) {
  constexpr int G = CIN / 32;
  constexpr int D = 9 * G * NT * 32 * 8;
  const int idx = blockIdx.x * 256 + threadIdx.x;
  if (idx >= D) return;
  const int v    = idx & 7;
  const int lane = (idx >> 3) & 31;
  const int f    = idx >> 8;
  const int nt   = f % NT;
  const int g    = (f / NT) % G;
  const int tap  = f / (NT * G);
  const int half = lane >> 4, col = lane & 15;
  const int o    = nt * 16 + col;
  const int c0   = g * 32 + kslot(2 * v, half);
  unsigned val = 0;
  if (o < COUT)
    val = pk_bf16(w[((size_t)o * CIN + c0) * 9 + tap],
                  w[((size_t)o * CIN + c0 + 1) * 9 + tap]);
  wp[idx] = val;
}

// ---------------------------------------------------------------------------
// 3x3 same-pad conv as implicit GEMM on packed layouts.
// src: bf16 NHWC [B,H,W,CIN]; wp: packed B fragments.
// OUT_NHWC: write bf16 NHWC [B,H,W,COUT] (conv1); else fp32 NCHW (+ACT).
// ---------------------------------------------------------------------------
template <int CIN, int COUT, int NT, int ACT, bool OUT_NHWC>
__global__ __launch_bounds__(256)
void conv3x3_wmma(const __bf16* __restrict__ src,
                  const unsigned* __restrict__ wp,
                  const float* __restrict__ bias, void* __restrict__ outv) {
  constexpr int G   = CIN / 32;
  constexpr int LSZ = OUT_NHWC ? 1024 : 256;
  __shared__ float lds[8 * LSZ];
  const int wid  = threadIdx.x >> 5;
  const int wave = blockIdx.x * 8 + wid;
  const int lane = threadIdx.x & 31;
  const int half = lane >> 4;
  const int col  = lane & 15;

  const int b   = wave / MT_PER_IMG;
  const int rem = wave % MT_PER_IMG;
  const int y   = rem / (Ww / 16);
  const int xb  = (rem % (Ww / 16)) * 16;

  __builtin_prefetch(wp, 0, 0);                     // global_prefetch_b8

  v8f acc[NT];
#pragma unroll
  for (int nt = 0; nt < NT; ++nt) acc[nt] = (v8f){};

  for (int tap = 0; tap < 9; ++tap) {
    const int dy = tap / 3 - 1, dx = tap % 3 - 1;
    const int yy = y + dy;
    const int xx = xb + col + dx;                   // this lane's A-row pixel
    const bool ok =
        ((unsigned)yy < (unsigned)Hh) && ((unsigned)xx < (unsigned)Ww);
    const unsigned amask = ok ? 0xFFFFFFFFu : 0u;   // lane-uniform over K
    const size_t pixoff =
        ((size_t)(b * Hh + min(max(yy, 0), Hh - 1)) * Ww +
         min(max(xx, 0), Ww - 1)) * CIN;

#pragma unroll
    for (int g = 0; g < G; ++g) {
      // A fragment: two contiguous 16B channel runs, then pad-mask.
      v8u au = mask8(load_frag_runs(src + pixoff + g * 32 + half * 8), amask);
      const v16bf a = __builtin_bit_cast(v16bf, au);

      const unsigned* wf = wp + ((size_t)(tap * G + g) * NT * 32 + lane) * 8;
#pragma unroll
      for (int nt = 0; nt < NT; ++nt) {             // reuse A across N-tiles
        const v16bf bb = __builtin_bit_cast(
            v16bf, *reinterpret_cast<const v8u*>(wf + (size_t)nt * 32 * 8));
        acc[nt] = __builtin_amdgcn_wmma_f32_16x16x32_bf16(
            false, a, false, bb, (short)0, acc[nt], false, false);
      }
    }
  }

  float* myl = lds + wid * LSZ;
  if constexpr (OUT_NHWC) {
    // feat epilogue: bias, transpose via LDS [ch64][px16], pack bf16 NHWC.
    __bf16* out = (__bf16*)outv;
#pragma unroll
    for (int nt = 0; nt < NT; ++nt) {
      const float bv = bias[nt * 16 + col];
#pragma unroll
      for (int r = 0; r < 8; ++r)
        myl[(nt * 16 + col) * 16 + r + half * 8] = acc[nt][r] + bv;
    }
    const int px = col, chb = half * 32;
    const size_t pbase = ((size_t)(b * Hh + y) * Ww + xb + px) * COUT;
#pragma unroll
    for (int q = 0; q < 4; ++q) {                   // 4x b128 store per lane
      uint4 u;
      u.x = pk_bf16(myl[(chb + q * 8 + 0) * 16 + px], myl[(chb + q * 8 + 1) * 16 + px]);
      u.y = pk_bf16(myl[(chb + q * 8 + 2) * 16 + px], myl[(chb + q * 8 + 3) * 16 + px]);
      u.z = pk_bf16(myl[(chb + q * 8 + 4) * 16 + px], myl[(chb + q * 8 + 5) * 16 + px]);
      u.w = pk_bf16(myl[(chb + q * 8 + 6) * 16 + px], myl[(chb + q * 8 + 7) * 16 + px]);
      *reinterpret_cast<uint4*>(out + pbase + chb + q * 8) = u;
    }
  } else {
    // offset/mod epilogue: bias (+2*sigmoid), coalesced fp32 NCHW stores.
    float* out = (float*)outv;
#pragma unroll
    for (int nt = 0; nt < NT; ++nt) {
      const int o    = nt * 16 + col;
      const float bv = bias[(o < COUT) ? o : 0];
#pragma unroll
      for (int r = 0; r < 8; ++r) {
        float v = acc[nt][r] + bv;
        if (ACT == 1) v = 2.f / (1.f + __expf(-v));
        myl[col * 16 + r + half * 8] = v;
      }
#pragma unroll
      for (int j = 0; j < 8; ++j) {
        const int ch = j * 2 + half;
        const int oo = nt * 16 + ch;
        if (NT * 16 == COUT || oo < COUT)
          out[(((size_t)b * COUT + oo) * Hh + y) * Ww + xb + col] =
              myl[ch * 16 + col];
      }
    }
  }
}

// ---------------------------------------------------------------------------
// DCNv2 contraction.  A = bilinear-sampled bf16-NHWC feat (modulation folded
// into the 4 corner weights), B = packed Wdcn fragments.  16 px x 64 out ch
// per wave; K = 9 taps x 2 groups of 32 channels.
// ---------------------------------------------------------------------------
__global__ __launch_bounds__(256)
void dcn_wmma(const __bf16* __restrict__ feat, const float* __restrict__ offs,
              const float* __restrict__ modm, const unsigned* __restrict__ wp,
              const float* __restrict__ bdcn, float* __restrict__ out) {
  __shared__ float lds[8 * 256];
  const int wid  = threadIdx.x >> 5;
  const int wave = blockIdx.x * 8 + wid;
  const int lane = threadIdx.x & 31;
  const int half = lane >> 4;
  const int col  = lane & 15;

  const int b   = wave / MT_PER_IMG;
  const int rem = wave % MT_PER_IMG;
  const int y   = rem / (Ww / 16);
  const int xb  = (rem % (Ww / 16)) * 16;
  const int x   = xb + col;                          // this lane's pixel

  __builtin_prefetch(wp, 0, 0);

  v8f acc[4];
#pragma unroll
  for (int nt = 0; nt < 4; ++nt) acc[nt] = (v8f){};

  const __bf16* fb = feat + (size_t)b * PLANE * Cc;  // NHWC base

  for (int tap = 0; tap < 9; ++tap) {
    const int ky = tap / 3 - 1, kx = tap % 3 - 1;
    const float dy = offs[(((size_t)b * 18 + 2 * tap)     * Hh + y) * Ww + x];
    const float dx = offs[(((size_t)b * 18 + 2 * tap + 1) * Hh + y) * Ww + x];
    const float mk = modm[(((size_t)b * 9 + tap) * Hh + y) * Ww + x];

    const float py = (float)(y + ky) + dy;
    const float px = (float)(x + kx) + dx;
    const float y0f = floorf(py), x0f = floorf(px);
    const float wy1 = py - y0f,   wx1 = px - x0f;
    const int y0 = (int)y0f, x0 = (int)x0f;
    const int y1 = y0 + 1,   x1 = x0 + 1;

    const bool vy0 = (y0 >= 0) && (y0 < Hh), vy1 = (y1 >= 0) && (y1 < Hh);
    const bool vx0 = (x0 >= 0) && (x0 < Ww), vx1 = (x1 >= 0) && (x1 < Ww);
    const int y0c = min(max(y0, 0), Hh - 1), y1c = min(max(y1, 0), Hh - 1);
    const int x0c = min(max(x0, 0), Ww - 1), x1c = min(max(x1, 0), Ww - 1);

    const float w00 = (vy0 && vx0) ? (1.f - wy1) * (1.f - wx1) * mk : 0.f;
    const float w01 = (vy0 && vx1) ? (1.f - wy1) * wx1         * mk : 0.f;
    const float w10 = (vy1 && vx0) ? wy1 * (1.f - wx1)         * mk : 0.f;
    const float w11 = (vy1 && vx1) ? wy1 * wx1                 * mk : 0.f;

    const size_t p00 = ((size_t)y0c * Ww + x0c) * Cc;
    const size_t p01 = ((size_t)y0c * Ww + x1c) * Cc;
    const size_t p10 = ((size_t)y1c * Ww + x0c) * Cc;
    const size_t p11 = ((size_t)y1c * Ww + x1c) * Cc;

#pragma unroll
    for (int g = 0; g < 2; ++g) {
      const int ch = g * 32 + half * 8;
      // 4 corners x two 16B channel runs = 8x global_load_b128.
      const v8u u00 = load_frag_runs(fb + p00 + ch);
      const v8u u01 = load_frag_runs(fb + p01 + ch);
      const v8u u10 = load_frag_runs(fb + p10 + ch);
      const v8u u11 = load_frag_runs(fb + p11 + ch);

      v16bf a;
#pragma unroll
      for (int v = 0; v < 8; ++v) {
        // bf16 -> f32 expansion via shift/and bit tricks.
        const float f00l = __builtin_bit_cast(float, u00[v] << 16);
        const float f01l = __builtin_bit_cast(float, u01[v] << 16);
        const float f10l = __builtin_bit_cast(float, u10[v] << 16);
        const float f11l = __builtin_bit_cast(float, u11[v] << 16);
        const float f00h = __builtin_bit_cast(float, u00[v] & 0xffff0000u);
        const float f01h = __builtin_bit_cast(float, u01[v] & 0xffff0000u);
        const float f10h = __builtin_bit_cast(float, u10[v] & 0xffff0000u);
        const float f11h = __builtin_bit_cast(float, u11[v] & 0xffff0000u);
        const float rl =
            fmaf(w11, f11l, fmaf(w10, f10l, fmaf(w01, f01l, w00 * f00l)));
        const float rh =
            fmaf(w11, f11h, fmaf(w10, f10h, fmaf(w01, f01h, w00 * f00h)));
        a[2 * v]     = (__bf16)rl;
        a[2 * v + 1] = (__bf16)rh;
      }

      const unsigned* wf = wp + ((size_t)(tap * 2 + g) * 4 * 32 + lane) * 8;
#pragma unroll
      for (int nt = 0; nt < 4; ++nt) {               // reuse A across N-tiles
        const v16bf bb = __builtin_bit_cast(
            v16bf, *reinterpret_cast<const v8u*>(wf + (size_t)nt * 32 * 8));
        acc[nt] = __builtin_amdgcn_wmma_f32_16x16x32_bf16(
            false, a, false, bb, (short)0, acc[nt], false, false);
      }
    }
  }

  // Epilogue: bias, LDS transpose, coalesced fp32 NCHW stores.
  float* myl = lds + wid * 256;
#pragma unroll
  for (int nt = 0; nt < 4; ++nt) {
    const float bv = bdcn[nt * 16 + col];
#pragma unroll
    for (int r = 0; r < 8; ++r)
      myl[col * 16 + r + half * 8] = acc[nt][r] + bv;
#pragma unroll
    for (int j = 0; j < 8; ++j) {
      const int ch = j * 2 + half;
      out[(((size_t)b * Cc + nt * 16 + ch) * Hh + y) * Ww + xb + col] =
          myl[ch * 16 + col];
    }
  }
}

extern "C" void kernel_launch(void* const* d_in, const int* in_sizes, int n_in,
                              void* d_out, int out_size, void* d_ws, size_t ws_size,
                              hipStream_t stream) {
  (void)in_sizes; (void)n_in; (void)out_size; (void)ws_size;

  const float* x_vq  = (const float*)d_in[0];
  const float* x_res = (const float*)d_in[1];
  const float* W1    = (const float*)d_in[2];
  const float* b1    = (const float*)d_in[3];
  const float* Woff  = (const float*)d_in[4];
  const float* boff  = (const float*)d_in[5];
  const float* Wmod  = (const float*)d_in[6];
  const float* bmod  = (const float*)d_in[7];
  const float* Wdcn  = (const float*)d_in[8];
  const float* bdcn  = (const float*)d_in[9];
  float* outp = (float*)d_out;

  // Workspace carve-up (all section sizes are 256B multiples).
  char* ws = (char*)d_ws;
  unsigned* catb  = (unsigned*)ws;                       // [B*H*W][64dw] bf16x2
  ws += (size_t)Bn * PLANE * 64 * 4;
  __bf16* featb   = (__bf16*)ws;                         // NHWC [B,H,W,64]
  ws += (size_t)Bn * PLANE * 64 * 2;
  float* offs     = (float*)ws;                          // NCHW [8,18,H,W]
  ws += (size_t)Bn * 18 * PLANE * 4;
  float* modm     = (float*)ws;                          // NCHW [8,9,H,W]
  ws += (size_t)Bn * 9 * PLANE * 4;
  unsigned* w1p   = (unsigned*)ws; ws += (size_t)9 * 4 * 4 * 256 * 4;
  unsigned* w2p   = (unsigned*)ws; ws += (size_t)9 * 2 * 2 * 256 * 4;
  unsigned* w3p   = (unsigned*)ws; ws += (size_t)9 * 2 * 1 * 256 * 4;
  unsigned* wdp   = (unsigned*)ws;

  // One-shot layout packs (streaming, L2-resident afterwards).
  pack_cat<<<(Bn * PLANE) / 256, 256, 0, stream>>>(x_vq, x_res, catb);
  pack_w<128, 64, 4><<<9 * 4 * 4, 256, 0, stream>>>(W1, w1p);
  pack_w< 64, 18, 2><<<9 * 2 * 2, 256, 0, stream>>>(Woff, w2p);
  pack_w< 64,  9, 1><<<9 * 2 * 1, 256, 0, stream>>>(Wmod, w3p);
  pack_w< 64, 64, 4><<<9 * 2 * 4, 256, 0, stream>>>(Wdcn, wdp);

  const int blocks = MTILES / 8;                         // 8 waves per block
  conv3x3_wmma<128, 64, 4, 0, true ><<<blocks, 256, 0, stream>>>(
      (const __bf16*)catb, w1p, b1, featb);
  conv3x3_wmma< 64, 18, 2, 0, false><<<blocks, 256, 0, stream>>>(
      featb, w2p, boff, offs);
  conv3x3_wmma< 64,  9, 1, 1, false><<<blocks, 256, 0, stream>>>(
      featb, w3p, bmod, modm);
  dcn_wmma<<<blocks, 256, 0, stream>>>(featb, offs, modm, wdp, bdcn, outp);
}
